// MambaLayer_82119774700186
// MI455X (gfx1250) — matrix operand order repaired and checked by reading, not compiled, not run on hardware
//
#include <hip/hip_runtime.h>

// ---------------------------------------------------------------- constants
constexpr int BATCH   = 8;
constexpr int DIM     = 256;
constexpr int DINNER  = 256;           // expand = 1
constexpr int LTOK    = 32 * 32 * 16;  // 16384 tokens per batch
constexpr int DTRANK  = 16;
constexpr int XDS     = 20;            // x_dbl row stride (18 padded to 20 for 16B align)
constexpr int NCHUNK  = 128;
constexpr int CLEN    = LTOK / NCHUNK; // 128

typedef __attribute__((ext_vector_type(16))) __bf16 v16bf;
typedef __attribute__((ext_vector_type(8)))  float  v8f;

union FragBF { v16bf v; uint4 q[2]; };

// CDNA5 async global->LDS path (ASYNCcnt), if the toolchain exposes it.
#if defined(__has_builtin)
#if __has_builtin(__builtin_amdgcn_global_load_async_to_lds_b128) && \
    __has_builtin(__builtin_amdgcn_s_wait_asynccnt)
#define USE_ASYNC_LDS 1
#endif
#endif
#ifndef USE_ASYNC_LDS
#define USE_ASYNC_LDS 0
#endif

typedef int v4i_ __attribute__((__vector_size__(16)));
typedef __attribute__((address_space(1))) v4i_ g_v4i;  // global int4
typedef __attribute__((address_space(3))) v4i_ l_v4i;  // LDS int4
typedef __attribute__((address_space(3))) char lds_char;
typedef __attribute__((address_space(1))) char glb_char;

static __device__ __forceinline__ unsigned short f2bf(float f) {
  union { float f; unsigned int u; } c; c.f = f;
  unsigned int r = c.u + 0x7FFFu + ((c.u >> 16) & 1u);
  return (unsigned short)(r >> 16);
}
static __device__ __forceinline__ float sigm(float x) {
  return 1.f / (1.f + __expf(-x));
}

// Cooperative 64KB tile fill: 256 threads x 16 x 16B.
static __device__ __forceinline__ void fill_tile(
    unsigned short* __restrict__ sDst, const unsigned short* __restrict__ gSrc, int t) {
#if USE_ASYNC_LDS
  lds_char* ld = (lds_char*)sDst;
  glb_char* gp = (glb_char*)gSrc;
  #pragma unroll
  for (int i = 0; i < 16; ++i) {
    int o = (t + 256 * i) * 16;
    __builtin_amdgcn_global_load_async_to_lds_b128((g_v4i*)(gp + o), (l_v4i*)(ld + o),
                                                   0, 0);
  }
#else
  uint4* d = (uint4*)sDst;
  const uint4* s = (const uint4*)gSrc;
  #pragma unroll
  for (int i = 0; i < 16; ++i) d[t + 256 * i] = s[t + 256 * i];
#endif
}
static __device__ __forceinline__ void tiles_ready() {
#if USE_ASYNC_LDS
  __builtin_amdgcn_s_wait_asynccnt(0);
#endif
  __syncthreads();
}

// ------------------------------------------------- 0) weights fp32 -> bf16
__global__ __launch_bounds__(256) void prep_weights(
    const float* __restrict__ ipw,   // (512,256)
    const float* __restrict__ opw,   // (256,256)
    unsigned short* __restrict__ wA,
    unsigned short* __restrict__ wO) {
  int idx = blockIdx.x * 256 + threadIdx.x;
  int n1 = 2 * DINNER * DIM;   // 131072
  int n2 = DIM * DINNER;       // 65536
  if (idx < n1) wA[idx] = f2bf(ipw[idx]);
  if (idx < n2) wO[idx] = f2bf(opw[idx]);
}

// ------------------------------------------------- 1) LayerNorm -> bf16, token-major
__global__ __launch_bounds__(256) void ln_kernel(
    const float* __restrict__ x,      // (B,256,L)
    const float* __restrict__ g, const float* __restrict__ bt,
    unsigned short* __restrict__ xn)  // (B,L,256) bf16
{
  __shared__ float red0[8][32], red1[8][32];
  __shared__ float mus[32], rss[32];
  __shared__ unsigned short tile[32][DIM + 8];

  const int tx = threadIdx.x, ty = threadIdx.y;     // (32,8)
  const int b  = blockIdx.y;
  const int l  = blockIdx.x * 32 + tx;

  float vals[32];
  float s = 0.f, s2 = 0.f;
  #pragma unroll 8
  for (int k = 0; k < 32; ++k) {
    int c = ty + 8 * k;
    float v = x[((size_t)b * DIM + c) * LTOK + l];
    vals[k] = v; s += v; s2 += v * v;
  }
  red0[ty][tx] = s; red1[ty][tx] = s2;
  __syncthreads();
  if (ty == 0) {
    float S = 0.f, S2 = 0.f;
    #pragma unroll
    for (int j = 0; j < 8; ++j) { S += red0[j][tx]; S2 += red1[j][tx]; }
    float mu = S * (1.f / DIM);
    float var = S2 * (1.f / DIM) - mu * mu;
    mus[tx] = mu; rss[tx] = rsqrtf(var + 1e-5f);
  }
  __syncthreads();
  float mu = mus[tx], rs = rss[tx];
  #pragma unroll 8
  for (int k = 0; k < 32; ++k) {
    int c = ty + 8 * k;
    float nv = (vals[k] - mu) * rs * g[c] + bt[c];
    tile[tx][c] = f2bf(nv);
  }
  __syncthreads();
  // contiguous 32*256 bf16 block: (B,L,256) rows l0..l0+31
  uint4* dst = (uint4*)(xn + ((size_t)b * LTOK + blockIdx.x * 32) * DIM);
  int t = ty * 32 + tx;
  #pragma unroll
  for (int i = 0; i < 4; ++i) {
    int u = t + 256 * i;            // 1024 uint4 total
    int row = u >> 5, col = u & 31; // 32 uint4 per row
    dst[u] = *(const uint4*)&tile[row][col * 8];
  }
}

// ------------------------------------------------- 2) in_proj GEMM (WMMA bf16)
// C[b,l,e] (e in 0..511) = sum_c xn[b,l,c] * W[e,c]
__global__ __launch_bounds__(256) void inproj_gemm(
    const unsigned short* __restrict__ xn,  // (B,L,256) bf16
    const unsigned short* __restrict__ wA,  // (512,256) bf16
    float* __restrict__ uz)                 // (B,L,512) f32
{
  extern __shared__ char smem_raw[];
  unsigned short* sA = (unsigned short*)smem_raw;   // 128x256
  unsigned short* sB = sA + 128 * 256;              // 128x256

  const int b  = blockIdx.z;
  const int l0 = blockIdx.x * 128;
  const int e0 = blockIdx.y * 128;
  const int t  = threadIdx.x;

  fill_tile(sA, xn + ((size_t)b * LTOK + l0) * DIM, t);
  fill_tile(sB, wA + (size_t)e0 * DIM, t);
  tiles_ready();

  const int wave = t >> 5, lane = t & 31;
  const int ll = lane & 15, lh = lane >> 4;

  // A fragments for this wave's 16-row M tile, all 8 K-blocks resident
  FragBF a[8];
  const int rowbase = (wave * 16 + ll) * DIM;
  #pragma unroll
  for (int kb = 0; kb < 8; ++kb) {
    int o = rowbase + kb * 32 + lh * 16;
    a[kb].q[0] = *(const uint4*)(sA + o);        // K +0..7   (or +16..23)
    a[kb].q[1] = *(const uint4*)(sA + o + 8);    // K +8..15  (or +24..31)
  }

  const int bbase = ll * DIM + lh * 16;
  #pragma unroll
  for (int nt = 0; nt < 8; ++nt) {
    const unsigned short* sBr = sB + nt * 16 * DIM + bbase;
    v8f c = {};
    FragBF bcur, bnxt;
    bcur.q[0] = ((const uint4*)sBr)[0];
    bcur.q[1] = ((const uint4*)sBr)[1];
    #pragma unroll
    for (int kb = 0; kb < 8; ++kb) {
      if (kb < 7) {   // double-buffer: prefetch next K-block's B fragment
        const uint4* pn = (const uint4*)(sBr + (kb + 1) * 32);
        bnxt.q[0] = pn[0]; bnxt.q[1] = pn[1];
      }
      c = __builtin_amdgcn_wmma_f32_16x16x32_bf16(false, a[kb].v, false, bcur.v,
                                                  (short)0, c, false, false);
      bcur = bnxt;
    }
    float* o = uz + ((size_t)b * LTOK + l0 + wave * 16 + lh * 8) * (2 * DINNER)
                  + (e0 + nt * 16 + ll);
    #pragma unroll
    for (int v = 0; v < 8; ++v) o[(size_t)v * (2 * DINNER)] = c[v];
  }
}

// ------------------------------------------------- 3) depthwise causal conv + SiLU
__global__ __launch_bounds__(256) void conv_silu_kernel(
    const float* __restrict__ uz,     // (B,L,512)
    const float* __restrict__ cw,     // (256,4)
    const float* __restrict__ cb,
    float* __restrict__ ucf,          // (B,L,256) f32
    unsigned short* __restrict__ ucb) // (B,L,256) bf16
{
  const int d = threadIdx.x, l = blockIdx.x, b = blockIdx.y;
  const float* ub = uz + (size_t)b * LTOK * (2 * DINNER);
  float4 w = ((const float4*)cw)[d];
  const float wt[4] = {w.x, w.y, w.z, w.w};
  float acc = cb[d];
  #pragma unroll
  for (int j = 0; j < 4; ++j) {
    int ls = l - 3 + j;
    if (ls >= 0) acc += wt[j] * ub[(size_t)ls * (2 * DINNER) + d];
  }
  float sv = acc * sigm(acc);
  size_t o = ((size_t)b * LTOK + l) * DINNER + d;
  ucf[o] = sv;
  ucb[o] = f2bf(sv);
}

// ------------------------------------------------- 4) x_proj: (tok,18) = uc x Wxp^T
__global__ __launch_bounds__(256) void xproj_kernel(
    const float* __restrict__ ucf,   // (B*L,256)
    const float* __restrict__ xpw,   // (18,256)
    float* __restrict__ xdbl)        // (B*L,20)
{
  const int wave = threadIdx.x >> 5, lane = threadIdx.x & 31;
  const size_t tok = (size_t)blockIdx.x * 8 + wave;
  const float* ur = ucf + tok * DINNER;
  float uv[8];
  #pragma unroll
  for (int j = 0; j < 8; ++j) uv[j] = ur[lane + 32 * j];
  for (int e = 0; e < DTRANK + 2; ++e) {
    const float* wr = xpw + e * DINNER;
    float acc = 0.f;
    #pragma unroll
    for (int j = 0; j < 8; ++j) acc += uv[j] * wr[lane + 32 * j];
    #pragma unroll
    for (int off = 16; off > 0; off >>= 1) acc += __shfl_xor(acc, off, 32);
    if (lane == 0) xdbl[tok * XDS + e] = acc;
  }
}

// ------------------------------------------------- 5) dt_proj + softplus + scan coeffs
__global__ __launch_bounds__(256) void dt_kernel(
    const float* __restrict__ xdbl,
    const float* __restrict__ dtw,   // (256,16)
    const float* __restrict__ dtb,
    const float* __restrict__ Alog,  // (256,1)
    const float* __restrict__ ucf,
    float* __restrict__ aarr,        // exp(dt*A)
    float* __restrict__ btarr)       // dt*B*u
{
  const int d = threadIdx.x, l = blockIdx.x, b = blockIdx.y;
  const size_t tok = (size_t)b * LTOK + l;
  const float* xd = xdbl + tok * XDS;
  const float4* xr = (const float4*)xd;
  const float4* wr = (const float4*)(dtw + d * DTRANK);
  float dot = dtb[d];
  #pragma unroll
  for (int q = 0; q < 4; ++q) {
    float4 xv = xr[q], wv = wr[q];
    dot += xv.x * wv.x + xv.y * wv.y + xv.z * wv.z + xv.w * wv.w;
  }
  float dtv = (dot > 20.f) ? dot : log1pf(__expf(dot));
  float Ad  = -__expf(Alog[d]);
  float Bv  = xd[DTRANK];
  size_t o  = tok * DINNER + d;
  aarr[o]   = __expf(dtv * Ad);
  btarr[o]  = dtv * Bv * ucf[o];
}

// ------------------------------------------------- 6) scan phase 1: per-chunk (P,Q)
__global__ __launch_bounds__(256) void scan1_kernel(
    const float* __restrict__ aarr, const float* __restrict__ btarr,
    float* __restrict__ cP, float* __restrict__ cQ)
{
  const int d = threadIdx.x, ch = blockIdx.x, b = blockIdx.y;
  size_t base = ((size_t)b * LTOK + (size_t)ch * CLEN) * DINNER + d;
  float P = 1.f, Q = 0.f;
  for (int i = 0; i < CLEN; ++i) {
    float av = aarr[base + (size_t)i * DINNER];
    float bv = btarr[base + (size_t)i * DINNER];
    P *= av; Q = av * Q + bv;
  }
  size_t o = ((size_t)b * NCHUNK + ch) * DINNER + d;
  cP[o] = P; cQ[o] = Q;
}

// ------------------------------------------------- 7) scan phase 2: chunk prefixes
__global__ __launch_bounds__(256) void scan2_kernel(
    const float* __restrict__ cP, const float* __restrict__ cQ,
    float* __restrict__ hin)
{
  const int d = threadIdx.x, b = blockIdx.x;
  float h = 0.f;
  for (int c = 0; c < NCHUNK; ++c) {
    size_t o = ((size_t)b * NCHUNK + c) * DINNER + d;
    hin[o] = h;
    h = cP[o] * h + cQ[o];
  }
}

// ------------------------------------------------- 8) scan phase 3: replay + gate
__global__ __launch_bounds__(256) void scan3_kernel(
    const float* __restrict__ aarr, const float* __restrict__ btarr,
    const float* __restrict__ hin,  const float* __restrict__ xdbl,
    const float* __restrict__ ucf,  const float* __restrict__ Dp,
    const float* __restrict__ uz,   unsigned short* __restrict__ ybf)
{
  const int d = threadIdx.x, ch = blockIdx.x, b = blockIdx.y;
  float h  = hin[((size_t)b * NCHUNK + ch) * DINNER + d];
  float Dv = Dp[d];
  for (int i = 0; i < CLEN; ++i) {
    size_t tok = (size_t)b * LTOK + (size_t)ch * CLEN + i;
    size_t o = tok * DINNER + d;
    h = aarr[o] * h + btarr[o];
    float Cv = xdbl[tok * XDS + DTRANK + 1];
    float y  = h * Cv + ucf[o] * Dv;
    float z  = uz[tok * (2 * DINNER) + DINNER + d];
    ybf[o] = f2bf(y * (z * sigm(z)));
  }
}

// ------------------------------------------------- 9) out_proj GEMM (WMMA bf16) + residual
__global__ __launch_bounds__(256) void outproj_gemm(
    const unsigned short* __restrict__ ybf,  // (B,L,256) bf16
    const unsigned short* __restrict__ wO,   // (256,256) bf16
    const float* __restrict__ xin,           // (B,256,L) residual
    float* __restrict__ out)                 // (B,256,L)
{
  extern __shared__ char smem_raw[];
  unsigned short* sA = (unsigned short*)smem_raw;
  unsigned short* sB = sA + 128 * 256;

  const int b  = blockIdx.z;
  const int l0 = blockIdx.x * 128;
  const int e0 = blockIdx.y * 128;
  const int t  = threadIdx.x;

  fill_tile(sA, ybf + ((size_t)b * LTOK + l0) * DINNER, t);
  fill_tile(sB, wO + (size_t)e0 * DINNER, t);
  tiles_ready();

  const int wave = t >> 5, lane = t & 31;
  const int ll = lane & 15, lh = lane >> 4;

  FragBF a[8];
  const int rowbase = (wave * 16 + ll) * DINNER;
  #pragma unroll
  for (int kb = 0; kb < 8; ++kb) {
    int o = rowbase + kb * 32 + lh * 16;
    a[kb].q[0] = *(const uint4*)(sA + o);
    a[kb].q[1] = *(const uint4*)(sA + o + 8);
  }

  const int bbase = ll * DINNER + lh * 16;
  #pragma unroll
  for (int nt = 0; nt < 8; ++nt) {
    const unsigned short* sBr = sB + nt * 16 * DINNER + bbase;
    v8f c = {};
    FragBF bcur, bnxt;
    bcur.q[0] = ((const uint4*)sBr)[0];
    bcur.q[1] = ((const uint4*)sBr)[1];
    #pragma unroll
    for (int kb = 0; kb < 8; ++kb) {
      if (kb < 7) {
        const uint4* pn = (const uint4*)(sBr + (kb + 1) * 32);
        bnxt.q[0] = pn[0]; bnxt.q[1] = pn[1];
      }
      c = __builtin_amdgcn_wmma_f32_16x16x32_bf16(false, a[kb].v, false, bcur.v,
                                                  (short)0, c, false, false);
      bcur = bnxt;
    }
    // D tile: lane holds output channel (e0+nt*16+ll), 8 consecutive tokens
    size_t idx = ((size_t)b * DIM + e0 + nt * 16 + ll) * LTOK
               + l0 + wave * 16 + lh * 8;
    const float4* rx = (const float4*)(xin + idx);
    float4 r0 = rx[0], r1 = rx[1];
    float4 w0 = make_float4(c[0] + r0.x, c[1] + r0.y, c[2] + r0.z, c[3] + r0.w);
    float4 w1 = make_float4(c[4] + r1.x, c[5] + r1.y, c[6] + r1.z, c[7] + r1.w);
    ((float4*)(out + idx))[0] = w0;
    ((float4*)(out + idx))[1] = w1;
  }
}

// ================================================================ launcher
extern "C" void kernel_launch(void* const* d_in, const int* in_sizes, int n_in,
                              void* d_out, int out_size, void* d_ws, size_t ws_size,
                              hipStream_t stream) {
  const float* x    = (const float*)d_in[0];
  const float* ln_g = (const float*)d_in[1];
  const float* ln_b = (const float*)d_in[2];
  const float* ipw  = (const float*)d_in[3];
  const float* cw   = (const float*)d_in[4];
  const float* cb   = (const float*)d_in[5];
  const float* xpw  = (const float*)d_in[6];
  const float* dtw  = (const float*)d_in[7];
  const float* dtb  = (const float*)d_in[8];
  const float* opw  = (const float*)d_in[9];
  const float* Dp   = (const float*)d_in[10];
  const float* Alog = (const float*)d_in[11];
  float* out = (float*)d_out;

  const size_t nTok = (size_t)BATCH * LTOK;
  char* ws = (char*)d_ws;
  size_t off = 0;
  auto carve = [&](size_t bytes) {
    size_t o = off; off = (off + bytes + 255) & ~(size_t)255; return o;
  };
  unsigned short* xn   = (unsigned short*)(ws + carve(nTok * DIM * 2));
  float*          uz   = (float*)         (ws + carve(nTok * 2 * DINNER * 4));
  float*          ucf  = (float*)         (ws + carve(nTok * DINNER * 4));
  unsigned short* ucb  = (unsigned short*)(ws + carve(nTok * DINNER * 2));
  float*          xdbl = (float*)         (ws + carve(nTok * XDS * 4));
  float*          aarr = (float*)         (ws + carve(nTok * DINNER * 4));
  float*          btar = (float*)         (ws + carve(nTok * DINNER * 4));
  unsigned short* ybf  = (unsigned short*)(ws + carve(nTok * DINNER * 2));
  unsigned short* wA   = (unsigned short*)(ws + carve((size_t)2 * DINNER * DIM * 2));
  unsigned short* wO   = (unsigned short*)(ws + carve((size_t)DIM * DINNER * 2));
  float*          cP   = (float*)         (ws + carve((size_t)BATCH * NCHUNK * DINNER * 4));
  float*          cQ   = (float*)         (ws + carve((size_t)BATCH * NCHUNK * DINNER * 4));
  float*          hin  = (float*)         (ws + carve((size_t)BATCH * NCHUNK * DINNER * 4));

  const size_t gemm_lds = 2 * 128 * 256 * sizeof(unsigned short); // 128 KB

  prep_weights<<<dim3((2 * DINNER * DIM + 255) / 256), dim3(256), 0, stream>>>(
      ipw, opw, wA, wO);
  ln_kernel<<<dim3(LTOK / 32, BATCH), dim3(32, 8), 0, stream>>>(
      x, ln_g, ln_b, xn);
  inproj_gemm<<<dim3(LTOK / 128, (2 * DINNER) / 128, BATCH), dim3(256), gemm_lds, stream>>>(
      xn, wA, uz);
  conv_silu_kernel<<<dim3(LTOK, BATCH), dim3(256), 0, stream>>>(
      uz, cw, cb, ucf, ucb);
  xproj_kernel<<<dim3((unsigned)(nTok / 8)), dim3(256), 0, stream>>>(
      ucf, xpw, xdbl);
  dt_kernel<<<dim3(LTOK, BATCH), dim3(256), 0, stream>>>(
      xdbl, dtw, dtb, Alog, ucf, aarr, btar);
  scan1_kernel<<<dim3(NCHUNK, BATCH), dim3(256), 0, stream>>>(
      aarr, btar, cP, cQ);
  scan2_kernel<<<dim3(BATCH), dim3(256), 0, stream>>>(
      cP, cQ, hin);
  scan3_kernel<<<dim3(NCHUNK, BATCH), dim3(256), 0, stream>>>(
      aarr, btar, hin, xdbl, ucf, Dp, uz, ybf);
  outproj_gemm<<<dim3(LTOK / 128, DIM / 128, BATCH), dim3(256), gemm_lds, stream>>>(
      ybf, wO, x, out);
}
